// TransfomerEncoder_9947144257849
// MI455X (gfx1250) — matrix operand order repaired and checked by reading, not compiled, not run on hardware
//
#include <hip/hip_runtime.h>
#include <hip/hip_bf16.h>

// ---------------- problem constants ----------------
#define BB    16
#define TT    512
#define CINC  300
#define EE    1024
#define HHN   8
#define DHD   128
#define NLAYER 3
#define MM    (BB*TT)      // 8192 rows of activations
#define LOUT  2048
#define KPAD  320          // CIN padded to multiple of 32
#define F3E   3072         // 3*E

typedef __bf16 bf16x16 __attribute__((ext_vector_type(16)));
typedef float  v8f     __attribute__((ext_vector_type(8)));
typedef int    v4i     __attribute__((ext_vector_type(4)));

// ---------------- optional async global->LDS path (gfx1250 ASYNCcnt) ----------
#if defined(__has_builtin)
#if __has_builtin(__builtin_amdgcn_global_load_async_to_lds_b128) && \
    __has_builtin(__builtin_amdgcn_s_wait_asynccnt)
#define USE_ASYNC_LDS 1
#endif
#endif

#ifdef USE_ASYNC_LDS
typedef __attribute__((address_space(1))) v4i g_v4i;   // global (prints as __device__)
typedef __attribute__((address_space(3))) v4i l_v4i;   // LDS (prints as __shared__)
__device__ __forceinline__ void async_copy16(const void* g, void* l) {
  // LDS[l] = MEM[g] (16 bytes per active lane), tracked with ASYNCcnt
  __builtin_amdgcn_global_load_async_to_lds_b128((g_v4i*)g, (l_v4i*)l, 0, 0);
}
__device__ __forceinline__ void async_wait0() {
  __builtin_amdgcn_s_wait_asynccnt(0);
}
#else
__device__ __forceinline__ void async_wait0() {}
#endif

// ---------------- WMMA helpers ----------------
__device__ __forceinline__ v8f wmma_bf16(bf16x16 a, bf16x16 b, v8f c) {
  // D = A(16x32 bf16) * B(32x16 bf16) + C(16x16 f32)
  return __builtin_amdgcn_wmma_f32_16x16x32_bf16(false, a, false, b, (short)0, c,
                                                 false, false);
}

union FragU { bf16x16 v; uint4 q[2]; };

// A fragment (16x32, row-major source with leading dim `ld` elements).
// Lane l: row = l&15 ; kbase = (l>>4)*8 ; elems 0..7 -> k=kb..kb+7, 8..15 -> k=kb+16..kb+23
__device__ __forceinline__ bf16x16 load_a_frag(const __bf16* base, int ld,
                                               int row0, int k0, int lane) {
  FragU u;
  const __bf16* p = base + (size_t)(row0 + (lane & 15)) * ld + k0 + ((lane >> 4) << 3);
  u.q[0] = *(const uint4*)p;
  u.q[1] = *(const uint4*)(p + 16);
  return u.v;
}

// B fragment (32x16). Element (k, n) lives at base[n*ld + k] (i.e. weight stored [N,K]).
// Lane l: col = l&15 ; kbase = (l>>4)*8 ; elems 0..7 -> k=kb..kb+7, 8..15 -> k=kb+16..kb+23
__device__ __forceinline__ bf16x16 load_b_frag(const __bf16* base, int ld,
                                               int k0, int col0, int lane) {
  FragU u;
  const __bf16* p = base + (size_t)(col0 + (lane & 15)) * ld + k0 + ((lane >> 4) << 3);
  u.q[0] = *(const uint4*)p;
  u.q[1] = *(const uint4*)(p + 16);
  return u.v;
}

// ---------------- small utility kernels ----------------
__global__ void k_f2bf(const float* __restrict__ s, __bf16* __restrict__ d, size_t n) {
  size_t i = (size_t)blockIdx.x * 256 + threadIdx.x;
  if (i < n) d[i] = (__bf16)s[i];
}

__global__ void k_f2bf_pad(const float* __restrict__ s, __bf16* __restrict__ d,
                           int rows, int cols, int pad) {
  size_t i = (size_t)blockIdx.x * 256 + threadIdx.x;
  size_t n = (size_t)rows * pad;
  if (i < n) {
    int c = (int)(i % pad);
    size_t r = i / pad;
    d[i] = (c < cols) ? (__bf16)s[r * cols + c] : (__bf16)0.0f;
  }
}

__device__ __forceinline__ float posenc(int t, int n) {
  int j = n >> 1;
  float ang = (float)t * __expf((float)(2 * j) * (-9.210340371976184f / (float)EE));
  return (n & 1) ? __cosf(ang) : __sinf(ang);
}

// ---------------- generic bf16 WMMA GEMM (double-buffered LDS) ----------------
// C[m,n] = sum_k A[m,k] * W[n,k] + bias[n]   (A: MxK bf16 rm, W: NxK bf16 rm)
// Block tile 128(M) x 64(N), K-step 32, 8 waves -> each wave 32x32 (2x2 WMMA tiles).
__global__ __launch_bounds__(256) void k_gemm_bf16(
    const __bf16* __restrict__ A, const __bf16* __restrict__ W,
    const float* __restrict__ bias, float* __restrict__ Cf, __bf16* __restrict__ Cb,
    int Ndim, int Kdim, int addPE) {
  __shared__ __align__(16) __bf16 sA[2][128 * 32];   // 2 x 8 KB
  __shared__ __align__(16) __bf16 sB[2][64 * 32];    // 2 x 4 KB

  const int m0 = blockIdx.y * 128, n0 = blockIdx.x * 64;
  const int tid = threadIdx.x, lane = tid & 31, w = tid >> 5;
  const int wm = w & 3, wn = w >> 2;           // 4 M-subtiles x 2 N-subtiles of waves

  auto stageA = [&](int buf, int k0) {
    int r = tid >> 1, hf = tid & 1;
    const uint4* src = (const uint4*)(A + (size_t)(m0 + r) * Kdim + k0 + hf * 16);
    uint4* dst = (uint4*)(sA[buf] + r * 32 + hf * 16);
#ifdef USE_ASYNC_LDS
    async_copy16(src, dst);
    async_copy16(src + 1, dst + 1);
#else
    dst[0] = src[0];
    dst[1] = src[1];
#endif
  };
  auto stageB = [&](int buf, int k0) {
    if (tid < 128) {
      int r = tid >> 1, hf = tid & 1;
      const uint4* src = (const uint4*)(W + (size_t)(n0 + r) * Kdim + k0 + hf * 16);
      uint4* dst = (uint4*)(sB[buf] + r * 32 + hf * 16);
#ifdef USE_ASYNC_LDS
      async_copy16(src, dst);
      async_copy16(src + 1, dst + 1);
#else
      dst[0] = src[0];
      dst[1] = src[1];
#endif
    }
  };

  v8f acc[2][2] = {};
  stageA(0, 0);
  stageB(0, 0);
  int cur = 0;

  for (int k0 = 0; k0 < Kdim; k0 += 32) {
    async_wait0();       // this wave's async copies into sX[cur] have landed
    __syncthreads();     // every wave's copies visible; prev buffer free

    if (k0 + 32 < Kdim) {           // kick off next tile into the other buffer
      stageA(cur ^ 1, k0 + 32);
      stageB(cur ^ 1, k0 + 32);
      __builtin_prefetch(A + (size_t)(m0 + (tid >> 1)) * Kdim + k0 + 64, 0, 1);
    }

    bf16x16 a0 = load_a_frag(sA[cur], 32, 32 * wm, 0, lane);
    bf16x16 a1 = load_a_frag(sA[cur], 32, 32 * wm + 16, 0, lane);
    bf16x16 b0 = load_b_frag(sB[cur], 32, 0, 32 * wn, lane);
    bf16x16 b1 = load_b_frag(sB[cur], 32, 0, 32 * wn + 16, lane);
    acc[0][0] = wmma_bf16(a0, b0, acc[0][0]);
    acc[0][1] = wmma_bf16(a0, b1, acc[0][1]);
    acc[1][0] = wmma_bf16(a1, b0, acc[1][0]);
    acc[1][1] = wmma_bf16(a1, b1, acc[1][1]);

    cur ^= 1;
    __syncthreads();     // all reads of old buffer complete before it is refilled
  }

  // epilogue: C/D layout -> lane holds col (lane&15), rows mb..mb+7 where mb = (lane<16?0:8)
#pragma unroll
  for (int mt = 0; mt < 2; ++mt)
#pragma unroll
    for (int nt = 0; nt < 2; ++nt) {
      int n = n0 + 32 * wn + 16 * nt + (lane & 15);
      int mb = m0 + 32 * wm + 16 * mt + ((lane >> 4) << 3);
      float bv = bias ? bias[n] : 0.0f;
#pragma unroll
      for (int i = 0; i < 8; ++i) {
        int m = mb + i;
        float v = acc[mt][nt][i] + bv;
        if (addPE) v += posenc(m & (TT - 1), n);
        if (Cf) Cf[(size_t)m * Ndim + n] = v;
        if (Cb) Cb[(size_t)m * Ndim + n] = (__bf16)v;
      }
    }
}

// ---------------- V transpose: vT[bh][dh][t] = V[b,t,h,dh] ----------------
__global__ void k_vt(const __bf16* __restrict__ qkv, __bf16* __restrict__ vT) {
  size_t idx = (size_t)blockIdx.x * 256 + threadIdx.x;  // 128*128*512 total
  int t  = (int)(idx & 511);
  int dh = (int)((idx >> 9) & 127);
  int bh = (int)(idx >> 16);
  int b = bh >> 3, hh = bh & 7;
  vT[idx] = qkv[(size_t)(b * TT + t) * F3E + 2 * EE + hh * DHD + dh];
}

// ---------------- attention: one block per (b, h, 16-query tile) ----------------
__global__ __launch_bounds__(256) void k_attn(const __bf16* __restrict__ qkv,
                                              const __bf16* __restrict__ vT,
                                              const int* __restrict__ token_count,
                                              __bf16* __restrict__ aout) {
  __shared__ __align__(16) __bf16 sQ[16 * 128];   // 4 KB
  __shared__ __align__(16) float  sS[16 * 512];   // 32 KB
  __shared__ __align__(16) __bf16 sP[16 * 512];   // 16 KB
  const int qb = blockIdx.x;             // 0..31 query tiles
  const int bh = blockIdx.y;             // 0..127
  const int b = bh >> 3, hh = bh & 7;
  const int q0 = qb * 16;
  const int tid = threadIdx.x, lane = tid & 31, w = tid >> 5;

  // stage Q tile 16x128 bf16
  {
    int r = tid >> 4, ch = tid & 15;
    const uint4* src =
        (const uint4*)(qkv + (size_t)(b * TT + q0 + r) * F3E + hh * DHD + ch * 8);
    uint4* dst = (uint4*)(sQ + r * 128 + ch * 8);
#ifdef USE_ASYNC_LDS
    async_copy16(src, dst);
#else
    *dst = *src;
#endif
  }
  async_wait0();
  __syncthreads();

  // S = Q K^T * scale ; wave w covers keys [64w, 64w+64)
  {
    v8f acc[4] = {};
    const __bf16* kbase = qkv + (size_t)b * TT * F3E + EE + hh * DHD;
#pragma unroll
    for (int kk = 0; kk < 4; ++kk) {
      int k0 = kk * 32;
      bf16x16 a = load_a_frag(sQ, 128, 0, k0, lane);
#pragma unroll
      for (int nt = 0; nt < 4; ++nt)
        acc[nt] = wmma_bf16(a, load_b_frag(kbase, F3E, k0, 64 * w + 16 * nt, lane),
                            acc[nt]);
    }
    const float scale = 0.08838834764831845f;  // 1/sqrt(128)
#pragma unroll
    for (int nt = 0; nt < 4; ++nt) {
      int nl = 64 * w + 16 * nt + (lane & 15);
      int mb = (lane >> 4) << 3;
#pragma unroll
      for (int i = 0; i < 8; ++i) sS[(mb + i) * 512 + nl] = acc[nt][i] * scale;
    }
  }
  __syncthreads();

  // masked row softmax (16 rows x 512), 16 threads per row
  {
    int r = tid >> 4, lg = tid & 15;
    int tc = token_count[b];
    float mx = -3.0e38f;
    for (int c = lg; c < 512; c += 16)
      mx = fmaxf(mx, (c < tc) ? sS[r * 512 + c] : -1e9f);
    for (int o = 8; o; o >>= 1) mx = fmaxf(mx, __shfl_xor(mx, o, 32));
    float sum = 0.0f;
    for (int c = lg; c < 512; c += 16) {
      float e = (c < tc) ? __expf(sS[r * 512 + c] - mx) : 0.0f;
      sS[r * 512 + c] = e;
      sum += e;
    }
    for (int o = 8; o; o >>= 1) sum += __shfl_xor(sum, o, 32);
    float inv = 1.0f / sum;
    for (int c = lg; c < 512; c += 16)
      sP[r * 512 + c] = (__bf16)(sS[r * 512 + c] * inv);
  }
  __syncthreads();

  // O = P V ; wave w produces dh columns [16w, 16w+16)
  {
    v8f acc = {};
    const __bf16* vbase = vT + (size_t)bh * DHD * TT;
    for (int kt = 0; kt < 16; ++kt) {
      int k0 = kt * 32;
      bf16x16 a = load_a_frag(sP, 512, 0, k0, lane);
      acc = wmma_bf16(a, load_b_frag(vbase, TT, k0, 16 * w, lane), acc);
    }
    int col = hh * DHD + 16 * w + (lane & 15);
    int mb = (lane >> 4) << 3;
#pragma unroll
    for (int i = 0; i < 8; ++i)
      aout[(size_t)(b * TT + q0 + mb + i) * EE + col] = (__bf16)acc[i];
  }
}

// ---------------- fused residual + LayerNorm (row per block) ----------------
__global__ __launch_bounds__(256) void k_ln(float* __restrict__ h,
                                            const float* __restrict__ o,
                                            const float* __restrict__ sc,
                                            const float* __restrict__ bt,
                                            __bf16* __restrict__ hb) {
  const int m = blockIdx.x, tid = threadIdx.x;
  __shared__ float red[16];
  float x[4], sum = 0.0f;
#pragma unroll
  for (int i = 0; i < 4; ++i) {
    int e = tid + i * 256;
    float v = h[(size_t)m * EE + e];
    if (o) v += o[(size_t)m * EE + e];
    x[i] = v;
    sum += v;
  }
  for (int off = 16; off; off >>= 1) sum += __shfl_xor(sum, off, 32);
  int w = tid >> 5, lane = tid & 31;
  if (lane == 0) red[w] = sum;
  __syncthreads();
  if (tid < 8) {
    float t = red[tid];
    for (int off = 4; off; off >>= 1) t += __shfl_xor(t, off, 32);
    if (tid == 0) red[8] = t;
  }
  __syncthreads();
  float mean = red[8] * (1.0f / EE);
  float sq = 0.0f;
#pragma unroll
  for (int i = 0; i < 4; ++i) { float d = x[i] - mean; sq += d * d; }
  for (int off = 16; off; off >>= 1) sq += __shfl_xor(sq, off, 32);
  if (lane == 0) red[w] = sq;
  __syncthreads();
  if (tid < 8) {
    float t = red[tid];
    for (int off = 4; off; off >>= 1) t += __shfl_xor(t, off, 32);
    if (tid == 0) red[9] = t;
  }
  __syncthreads();
  float inv = rsqrtf(red[9] * (1.0f / EE) + 1e-5f);
#pragma unroll
  for (int i = 0; i < 4; ++i) {
    int e = tid + i * 256;
    float v = (x[i] - mean) * inv * sc[e] + bt[e];
    h[(size_t)m * EE + e] = v;
    if (hb) hb[(size_t)m * EE + e] = (__bf16)v;
  }
}

// ---------------- tiny FFN (DFF = 3), fp32 ----------------
__global__ __launch_bounds__(256) void k_ffn(const float* __restrict__ h,
                                             const float* __restrict__ W1,
                                             const float* __restrict__ b1,
                                             const float* __restrict__ W2,
                                             const float* __restrict__ b2,
                                             float* __restrict__ o) {
  const int m = blockIdx.x, tid = threadIdx.x;
  __shared__ float red[8][3];
  __shared__ float ff[3];
  float p0 = 0.f, p1 = 0.f, p2 = 0.f;
  for (int k = tid; k < EE; k += 256) {
    float xv = h[(size_t)m * EE + k];
    p0 += xv * W1[k];
    p1 += xv * W1[EE + k];
    p2 += xv * W1[2 * EE + k];
  }
  for (int off = 16; off; off >>= 1) {
    p0 += __shfl_xor(p0, off, 32);
    p1 += __shfl_xor(p1, off, 32);
    p2 += __shfl_xor(p2, off, 32);
  }
  int w = tid >> 5, lane = tid & 31;
  if (lane == 0) { red[w][0] = p0; red[w][1] = p1; red[w][2] = p2; }
  __syncthreads();
  if (tid < 3) {
    float t = 0.f;
    for (int i = 0; i < 8; ++i) t += red[i][tid];
    ff[tid] = fmaxf(t + b1[tid], 0.0f);
  }
  __syncthreads();
  float f0 = ff[0], f1 = ff[1], f2 = ff[2];
  for (int e = tid; e < EE; e += 256)
    o[(size_t)m * EE + e] = b2[e] + f0 * W2[e * 3] + f1 * W2[e * 3 + 1] + f2 * W2[e * 3 + 2];
}

// ---------------- length regulator ----------------
__global__ void k_csum(const int* __restrict__ dur, int* __restrict__ csum) {
  int b = threadIdx.x;
  if (b < BB) {
    int a = 0;
    for (int t = 0; t < TT; ++t) { a += dur[b * TT + t]; csum[b * TT + t] = a; }
  }
}

__global__ __launch_bounds__(256) void k_gather(const float* __restrict__ h,
                                                const int* __restrict__ csum,
                                                float* __restrict__ out) {
  int id = blockIdx.x;           // B*LOUT blocks
  int b = id >> 11, ll = id & (LOUT - 1);
  // searchsorted(csum, ll, 'right'): first i with csum[i] > ll
  int lo = 0, hi = TT;
  while (lo < hi) {
    int mid = (lo + hi) >> 1;
    if (csum[b * TT + mid] <= ll) lo = mid + 1; else hi = mid;
  }
  int idx = lo < TT ? lo : TT - 1;
  const float4* src = (const float4*)(h + (size_t)(b * TT + idx) * EE) + threadIdx.x;
  float4* dst = (float4*)(out + (size_t)id * EE) + threadIdx.x;
  *dst = *src;
}

// ---------------- host driver ----------------
extern "C" void kernel_launch(void* const* d_in, const int* in_sizes, int n_in,
                              void* d_out, int out_size, void* d_ws, size_t ws_size,
                              hipStream_t stream) {
  const float* x     = (const float*)d_in[0];
  const int* tok_cnt = (const int*)d_in[1];
  const int* tok_dur = (const int*)d_in[2];
  const float* w_tin = (const float*)d_in[4];
  const float* b_tin = (const float*)d_in[5];
  const float* Wqkv  = (const float*)d_in[6];
  const float* bqkv  = (const float*)d_in[7];
  const float* Wo    = (const float*)d_in[8];
  const float* bo    = (const float*)d_in[9];
  const float* W1    = (const float*)d_in[10];
  const float* b1    = (const float*)d_in[11];
  const float* W2    = (const float*)d_in[12];
  const float* b2    = (const float*)d_in[13];
  const float* ln1s  = (const float*)d_in[14];
  const float* ln1b  = (const float*)d_in[15];
  const float* ln2s  = (const float*)d_in[16];
  const float* ln2b  = (const float*)d_in[17];
  const float* lnfs  = (const float*)d_in[18];
  const float* lnfb  = (const float*)d_in[19];

  // workspace carve-up (256B aligned)
  size_t off = 0;
  auto carve = [&](size_t bytes) -> char* {
    char* p = (char*)d_ws + off;
    off += (bytes + 255) & ~(size_t)255;
    return p;
  };
  __bf16* xpad    = (__bf16*)carve((size_t)MM * KPAD * 2);
  __bf16* wtin_bf = (__bf16*)carve((size_t)EE * KPAD * 2);
  __bf16* wqkv_bf = (__bf16*)carve((size_t)NLAYER * F3E * EE * 2);
  __bf16* wo_bf   = (__bf16*)carve((size_t)NLAYER * EE * EE * 2);
  float*  hbuf    = (float*)carve((size_t)MM * EE * 4);
  __bf16* hbf     = (__bf16*)carve((size_t)MM * EE * 2);
  __bf16* qkv_bf  = (__bf16*)carve((size_t)MM * F3E * 2);
  __bf16* vT      = (__bf16*)carve((size_t)BB * HHN * DHD * TT * 2);
  __bf16* aout_bf = (__bf16*)carve((size_t)MM * EE * 2);
  float*  obuf    = (float*)carve((size_t)MM * EE * 4);
  int*    csum    = (int*)carve((size_t)BB * TT * 4);
  (void)ws_size; (void)n_in; (void)in_sizes; (void)out_size;

  auto blks = [](size_t n) { return (unsigned)((n + 255) / 256); };

  // weight / input conversion to bf16
  k_f2bf_pad<<<blks((size_t)MM * KPAD), 256, 0, stream>>>(x, xpad, MM, CINC, KPAD);
  k_f2bf_pad<<<blks((size_t)EE * KPAD), 256, 0, stream>>>(w_tin, wtin_bf, EE, CINC, KPAD);
  k_f2bf<<<blks((size_t)NLAYER * F3E * EE), 256, 0, stream>>>(Wqkv, wqkv_bf,
                                                              (size_t)NLAYER * F3E * EE);
  k_f2bf<<<blks((size_t)NLAYER * EE * EE), 256, 0, stream>>>(Wo, wo_bf,
                                                             (size_t)NLAYER * EE * EE);

  // input projection + pos-enc
  k_gemm_bf16<<<dim3(EE / 64, MM / 128), 256, 0, stream>>>(
      xpad, wtin_bf, b_tin, hbuf, nullptr, EE, KPAD, /*addPE=*/1);
  k_f2bf<<<blks((size_t)MM * EE), 256, 0, stream>>>(hbuf, hbf, (size_t)MM * EE);

  for (int l = 0; l < NLAYER; ++l) {
    // qkv projection -> bf16
    k_gemm_bf16<<<dim3(F3E / 64, MM / 128), 256, 0, stream>>>(
        hbf, wqkv_bf + (size_t)l * F3E * EE, bqkv + (size_t)l * F3E,
        nullptr, qkv_bf, F3E, EE, 0);
    // V transpose for contiguous B fragments in P*V
    k_vt<<<blks((size_t)BB * HHN * DHD * TT), 256, 0, stream>>>(qkv_bf, vT);
    // attention
    k_attn<<<dim3(TT / 16, BB * HHN), 256, 0, stream>>>(qkv_bf, vT, tok_cnt, aout_bf);
    // output projection
    k_gemm_bf16<<<dim3(EE / 64, MM / 128), 256, 0, stream>>>(
        aout_bf, wo_bf + (size_t)l * EE * EE, bo + (size_t)l * EE,
        obuf, nullptr, EE, EE, 0);
    // h = LN(h + o)
    k_ln<<<MM, 256, 0, stream>>>(hbuf, obuf, ln1s + (size_t)l * EE,
                                 ln1b + (size_t)l * EE, hbf);
    // FFN (DFF=3)
    k_ffn<<<MM, 256, 0, stream>>>(hbuf, W1 + (size_t)l * 3 * EE, b1 + (size_t)l * 3,
                                  W2 + (size_t)l * EE * 3, b2 + (size_t)l * EE, obuf);
    // h = LN(h + ff)
    k_ln<<<MM, 256, 0, stream>>>(hbuf, obuf, ln2s + (size_t)l * EE,
                                 ln2b + (size_t)l * EE, hbf);
  }

  // final LN
  k_ln<<<MM, 256, 0, stream>>>(hbuf, nullptr, lnfs, lnfb, nullptr);

  // length regulator
  k_csum<<<1, 32, 0, stream>>>(tok_dur, csum);
  k_gather<<<BB * LOUT, 256, 0, stream>>>(hbuf, csum, (float*)d_out);
}
